// AdaptiveSoftmax_85839216378241
// MI455X (gfx1250) — matrix operand order repaired and checked by reading, not compiled, not run on hardware
//
#include <hip/hip_runtime.h>
#include <hip/hip_bf16.h>

// ---------------------------------------------------------------------------
// Adaptive softmax loss for MI455X (gfx1250, wave32, WMMA).
//   1) proj_kernel     : y_i = hidden @ P_i^T   (bf16 WMMA, packed b128 stores)
//   2) lse_gemm_kernel : logits = y_i @ W_i^T + b_i, fused two-pass logsumexp
//   3) lse_reduce      : combine per-block (max,sumexp) partials -> logZ
//   4) gather_loss     : per-token wave dot for the gathered logits
//   5) mean_kernel     : final mean(-logprob)
// NOTE: token-tile loops are '#pragma unroll 1' — full unroll replicates the
// 32-VGPR accumulator 8x, overflows into the VGPR-MSB space and spills the
// LDS A-fragments to scratch (seen in round-2 asm).
// ---------------------------------------------------------------------------

typedef __attribute__((ext_vector_type(16))) __bf16 v16bf;
typedef __attribute__((ext_vector_type(8)))  float  v8f;
typedef __attribute__((ext_vector_type(4)))  unsigned int u32x4;

union FragA  { v16bf v; u32x4 q[2]; };
union PackBF { u32x4 q; __bf16 h[8]; };

#define TOKENS 1024      // T*B
#define ROWS   64        // W rows staged per workgroup
#define DPROJ  1024
#define NEGBIG (-1.0e30f)

// ---------------------------------------------------------------------------
// GEMM + fused logsumexp.  M = W rows, N = tokens, K = embed dim (mult of 32).
// A (W block) staged in LDS as bf16; bias staged with -inf sentinel padding.
// B (y) read from global as [token][K] bf16.
// ---------------------------------------------------------------------------
template <int K, int KREAL>
__global__ __launch_bounds__(256)
void lse_gemm_kernel(const float* __restrict__ W, const float* __restrict__ bias,
                     const __bf16* __restrict__ yB, int nrows,
                     float2* __restrict__ partials, int partial_base)
{
    constexpr int PITCH = K + 8;              // +16B pad: keeps 16B align, spreads banks
    extern __shared__ __bf16 lds[];           // [ROWS][PITCH] + 64 floats of bias
    float* ldsB = (float*)(lds + ROWS * PITCH);
    const int tid     = threadIdx.x;
    const int rowbase = blockIdx.x * ROWS;

    // ---- stage W block -> LDS as bf16 (zero-pad rows >= nrows, k >= KREAL) ----
    for (int idx = tid * 4; idx < ROWS * K; idx += 256 * 4) {
        const int r  = idx / K;
        const int k  = idx - r * K;
        const int gr = rowbase + r;
        float4 w4 = make_float4(0.f, 0.f, 0.f, 0.f);
        if (gr < nrows && k < KREAL)
            w4 = *(const float4*)(W + (size_t)gr * KREAL + k);
        __bf16* dst = &lds[r * PITCH + k];
        dst[0] = (__bf16)w4.x; dst[1] = (__bf16)w4.y;
        dst[2] = (__bf16)w4.z; dst[3] = (__bf16)w4.w;
    }
    // ---- stage bias with -inf sentinel: makes padded rows vanish branchlessly
    if (tid < ROWS) {
        const int gr = rowbase + tid;
        ldsB[tid] = (gr < nrows) ? bias[gr] : -__builtin_inff();
    }
    __syncthreads();

    const int  wave  = tid >> 5;
    const int  lane  = tid & 31;
    const int  ln    = lane & 15;
    const bool hi    = lane >= 16;
    const int  klo   = hi ? 16 : 0;   // A-frag: lanes 16-31 hold K+16..31
    const int  koff  = hi ? 16 : 0;   // B-frag: lanes 16-31 hold K+16..31
    const int  mrow0 = hi ? 8  : 0;   // C-frag: lanes 16-31 hold M = j+8

#pragma unroll 1                      // keep one acc set live: no spills
    for (int tt = wave; tt < TOKENS / 16; tt += 8) {
        const int tok = tt * 16 + ln;
        const __bf16* yrow = yB + (size_t)tok * K + koff;
        v8f acc[4] = {};

#pragma unroll 1
        for (int k = 0; k < K; k += 32) {
            v16bf bf = *(const v16bf*)(yrow + k);
            FragA fa[4];
#pragma unroll
            for (int rt = 0; rt < 4; ++rt) {
                const __bf16* ap = &lds[(rt * 16 + ln) * PITCH + k + klo];
                fa[rt].q[0] = *(const u32x4*)(ap);        // K = k+klo    .. +7
                fa[rt].q[1] = *(const u32x4*)(ap + 8);    // K = k+8+klo  .. +7
            }
#pragma unroll
            for (int rt = 0; rt < 4; ++rt)
                acc[rt] = __builtin_amdgcn_wmma_f32_16x16x32_bf16(
                    false, fa[rt].v, false, bf, (short)0, acc[rt], false, false);
        }

        // ---- two-pass logsumexp over this block's rows for my token ----
        float m = NEGBIG;                    // finite floor: no NaN from empty halves
#pragma unroll
        for (int rt = 0; rt < 4; ++rt) {
#pragma unroll
            for (int j = 0; j < 8; ++j) {
                const float v = acc[rt][j] + ldsB[rt * 16 + mrow0 + j];
                acc[rt][j] = v;              // padded rows become -inf
                m = fmaxf(m, v);
            }
        }
        float s = 0.f;
#pragma unroll
        for (int rt = 0; rt < 4; ++rt)
#pragma unroll
            for (int j = 0; j < 8; ++j)
                s += __expf(acc[rt][j] - m); // exp(-inf - finite) == 0

        // merge the two half-wave row groups (same token, rows j vs j+8)
        const float om = __shfl_xor(m, 16, 32);
        const float os = __shfl_xor(s, 16, 32);
        const float nm = fmaxf(m, om);
        s = s * __expf(m - nm) + os * __expf(om - nm);
        m = nm;
        if (!hi)
            partials[(size_t)(partial_base + blockIdx.x) * TOKENS + tok] =
                make_float2(m, s);
    }
}

// ---------------------------------------------------------------------------
// Projection: y = hidden @ P^T, output bf16 [token][kp_out] (zero-padded cols).
// One lane's 8 C rows are 8 consecutive e's of one token -> one b128 store.
// ---------------------------------------------------------------------------
__global__ __launch_bounds__(256)
void proj_kernel(const float* __restrict__ hidden, const float* __restrict__ P,
                 __bf16* __restrict__ y, int n_e, int kp_out)
{
    constexpr int K = DPROJ;
    constexpr int PITCH = K + 8;
    extern __shared__ __bf16 lds[];
    const int tid     = threadIdx.x;
    const int rowbase = blockIdx.x * ROWS;

    for (int idx = tid * 4; idx < ROWS * K; idx += 256 * 4) {
        const int r  = idx / K;
        const int k  = idx - r * K;
        const int gr = rowbase + r;
        float4 w4 = make_float4(0.f, 0.f, 0.f, 0.f);
        if (gr < n_e)
            w4 = *(const float4*)(P + (size_t)gr * K + k);
        __bf16* dst = &lds[r * PITCH + k];
        dst[0] = (__bf16)w4.x; dst[1] = (__bf16)w4.y;
        dst[2] = (__bf16)w4.z; dst[3] = (__bf16)w4.w;
    }
    __syncthreads();

    const int  wave  = tid >> 5;
    const int  lane  = tid & 31;
    const int  ln    = lane & 15;
    const bool hi    = lane >= 16;
    const int  klo   = hi ? 16 : 0;
    const int  koff  = hi ? 16 : 0;
    const int  mrow0 = hi ? 8  : 0;

#pragma unroll 1                      // keep one acc set live: no spills
    for (int tt = wave; tt < TOKENS / 16; tt += 8) {
        const int tok = tt * 16 + ln;
        const float* hrow = hidden + (size_t)tok * K + koff;
        v8f acc[4] = {};

#pragma unroll 1
        for (int k = 0; k < K; k += 32) {
            v16bf bf;
#pragma unroll
            for (int x = 0; x < 16; ++x)
                bf[x] = (__bf16)hrow[k + x];
            FragA fa[4];
#pragma unroll
            for (int rt = 0; rt < 4; ++rt) {
                const __bf16* ap = &lds[(rt * 16 + ln) * PITCH + k + klo];
                fa[rt].q[0] = *(const u32x4*)(ap);
                fa[rt].q[1] = *(const u32x4*)(ap + 8);
            }
#pragma unroll
            for (int rt = 0; rt < 4; ++rt)
                acc[rt] = __builtin_amdgcn_wmma_f32_16x16x32_bf16(
                    false, fa[rt].v, false, bf, (short)0, acc[rt], false, false);
        }

        // packed bf16 stores: 8 consecutive e's per lane -> one b128
#pragma unroll
        for (int rt = 0; rt < 4; ++rt) {
            const int e0 = rowbase + rt * 16 + mrow0;   // multiple of 8
            if (e0 < kp_out) {                          // whole chunk in/out
                PackBF pk;
                const bool valid = e0 < n_e;            // n_e is a multiple of 8
#pragma unroll
                for (int j = 0; j < 8; ++j)
                    pk.h[j] = valid ? (__bf16)acc[rt][j] : (__bf16)0.f;
                *(u32x4*)(y + (size_t)tok * kp_out + e0) = pk.q;
            }
        }
    }
}

// ---------------------------------------------------------------------------
// Combine per-block (max, sumexp) partials -> logZ per token.
// ---------------------------------------------------------------------------
__global__ __launch_bounds__(256)
void lse_reduce_kernel(const float2* __restrict__ partials, int nblocks,
                       float* __restrict__ logZ)
{
    const int tok = blockIdx.x * 256 + threadIdx.x;
    if (tok >= TOKENS) return;
    float m = NEGBIG, s = 0.f;
    for (int b = 0; b < nblocks; ++b) {
        const float2 p  = partials[(size_t)b * TOKENS + tok];
        const float  nm = fmaxf(m, p.x);
        s = s * __expf(m - nm) + p.y * __expf(p.x - nm);
        m = nm;
    }
    logZ[tok] = m + __logf(s);
}

// ---------------------------------------------------------------------------
// Per-token gathered logits (one wave32 per token) + per-token logprob.
// ---------------------------------------------------------------------------
__device__ __forceinline__ float wave_dot(const __bf16* __restrict__ y,
                                          const float* __restrict__ w,
                                          int K, int lane)
{
    float s = 0.f;
    for (int k = lane; k < K; k += 32)
        s += (float)y[k] * w[k];
#pragma unroll
    for (int off = 16; off; off >>= 1)
        s += __shfl_xor(s, off, 32);
    return s;
}

__global__ __launch_bounds__(256)
void gather_loss_kernel(const int* __restrict__ target,
                        const __bf16* __restrict__ y0, const __bf16* __restrict__ y1,
                        const __bf16* __restrict__ y2, const __bf16* __restrict__ y3,
                        const float* __restrict__ W0, const float* __restrict__ b0,
                        const float* __restrict__ W1, const float* __restrict__ b1,
                        const float* __restrict__ W2, const float* __restrict__ b2,
                        const float* __restrict__ W3, const float* __restrict__ b3,
                        const float* __restrict__ cw, const float* __restrict__ cb,
                        const float* __restrict__ logZ,   // [4][TOKENS]
                        float* __restrict__ loss)
{
    const int gtid = blockIdx.x * 256 + threadIdx.x;
    const int tok  = gtid >> 5;
    const int lane = threadIdx.x & 31;
    if (tok >= TOKENS) return;

    const int tgt = target[tok];
    int ci, l_idx;
    if      (tgt < 20000)  { ci = 0; l_idx = 0;      }
    else if (tgt < 40000)  { ci = 1; l_idx = 20000;  }
    else if (tgt < 200000) { ci = 2; l_idx = 40000;  }
    else                   { ci = 3; l_idx = 200000; }
    const int local = tgt - l_idx;

    float lp;
    if (ci == 0) {
        const float d = wave_dot(y0 + (size_t)tok * DPROJ,
                                 W0 + (size_t)local * DPROJ, DPROJ, lane);
        lp = (d + b0[local]) - logZ[tok];
    } else {
        const float dh = wave_dot(y0 + (size_t)tok * DPROJ,
                                  cw + (size_t)(ci - 1) * DPROJ, DPROJ, lane);
        const float head_lp = (dh + cb[ci - 1]) - logZ[tok];

        const __bf16* yt; const float* Wt; const float* bt; int Kt, Kpt;
        if (ci == 1)      { yt = y1; Wt = W1; bt = b1; Kt = 256; Kpt = 256; }
        else if (ci == 2) { yt = y2; Wt = W2; bt = b2; Kt = 64;  Kpt = 64;  }
        else              { yt = y3; Wt = W3; bt = b3; Kt = 16;  Kpt = 32;  }

        const float dt = wave_dot(yt + (size_t)tok * Kpt,
                                  Wt + (size_t)local * Kt, Kt, lane);
        const float tail_lp = (dt + bt[local]) - logZ[ci * TOKENS + tok];
        lp = head_lp + tail_lp;
    }
    if (lane == 0) loss[tok] = lp;
}

__global__ __launch_bounds__(256)
void mean_kernel(const float* __restrict__ loss, float* __restrict__ out)
{
    __shared__ float red[256];
    float s = 0.f;
    for (int i = threadIdx.x; i < TOKENS; i += 256) s += loss[i];
    red[threadIdx.x] = s;
    __syncthreads();
    for (int w = 128; w > 0; w >>= 1) {
        if (threadIdx.x < w) red[threadIdx.x] += red[threadIdx.x + w];
        __syncthreads();
    }
    if (threadIdx.x == 0) out[0] = -red[0] / (float)TOKENS;
}

// ---------------------------------------------------------------------------
extern "C" void kernel_launch(void* const* d_in, const int* in_sizes, int n_in,
                              void* d_out, int out_size, void* d_ws, size_t ws_size,
                              hipStream_t stream)
{
    (void)in_sizes; (void)n_in; (void)out_size; (void)ws_size;
    const float* hidden = (const float*)d_in[0];
    const int*   target = (const int*)d_in[1];
    const float* W0 = (const float*)d_in[2];
    const float* b0 = (const float*)d_in[3];
    const float* P0 = (const float*)d_in[4];
    const float* W1 = (const float*)d_in[5];
    const float* b1 = (const float*)d_in[6];
    const float* P1 = (const float*)d_in[7];
    const float* W2 = (const float*)d_in[8];
    const float* b2 = (const float*)d_in[9];
    const float* P2 = (const float*)d_in[10];
    const float* W3 = (const float*)d_in[11];
    const float* b3 = (const float*)d_in[12];
    const float* P3 = (const float*)d_in[13];
    const float* cw = (const float*)d_in[14];
    const float* cb = (const float*)d_in[15];

    // ---- workspace carve ----
    char*  ws  = (char*)d_ws;
    size_t off = 0;
    auto carve = [&](size_t bytes) {
        void* p = (void*)(ws + off);
        off += (bytes + 255) & ~(size_t)255;
        return p;
    };
    const int NB_HEAD0 = (20000  + ROWS - 1) / ROWS;  // 313
    const int NB_HEAD  = NB_HEAD0 + 1;                // +1 block for cluster rows
    const int NB1      = (20000  + ROWS - 1) / ROWS;  // 313
    const int NB2      = (160000 + ROWS - 1) / ROWS;  // 2500
    const int NB3      = (67735  + ROWS - 1) / ROWS;  // 1059

    __bf16* y0 = (__bf16*)carve((size_t)TOKENS * 1024 * sizeof(__bf16));
    __bf16* y1 = (__bf16*)carve((size_t)TOKENS * 256  * sizeof(__bf16));
    __bf16* y2 = (__bf16*)carve((size_t)TOKENS * 64   * sizeof(__bf16));
    __bf16* y3 = (__bf16*)carve((size_t)TOKENS * 32   * sizeof(__bf16));
    float2* pH = (float2*)carve((size_t)NB_HEAD * TOKENS * sizeof(float2));
    float2* p1 = (float2*)carve((size_t)NB1     * TOKENS * sizeof(float2));
    float2* p2 = (float2*)carve((size_t)NB2     * TOKENS * sizeof(float2));
    float2* p3 = (float2*)carve((size_t)NB3     * TOKENS * sizeof(float2));
    float*  logZ = (float*)carve((size_t)4 * TOKENS * sizeof(float));
    float*  loss = (float*)carve((size_t)TOKENS * sizeof(float));

    // LDS: bf16 tile + 64-float bias block
    auto shbytes = [](int K) {
        return (size_t)ROWS * (K + 8) * sizeof(__bf16) + ROWS * sizeof(float);
    };

    // allow >64KB dynamic LDS for the K=1024 kernels (320KB/WGP on CDNA5)
    hipFuncSetAttribute(reinterpret_cast<const void*>(&proj_kernel),
                        hipFuncAttributeMaxDynamicSharedMemorySize, (int)shbytes(1024));
    hipFuncSetAttribute(reinterpret_cast<const void*>(&lse_gemm_kernel<1024, 1024>),
                        hipFuncAttributeMaxDynamicSharedMemorySize, (int)shbytes(1024));

    // 1) projections (bf16 outputs, K padded to multiple of 32)
    proj_kernel<<<16, 256, shbytes(1024), stream>>>(hidden, P0, y0, 1024, 1024);
    proj_kernel<<<4,  256, shbytes(1024), stream>>>(hidden, P1, y1, 256,  256);
    proj_kernel<<<1,  256, shbytes(1024), stream>>>(hidden, P2, y2, 64,   64);
    proj_kernel<<<1,  256, shbytes(1024), stream>>>(hidden, P3, y3, 16,   32);

    // 2) GEMM + fused logsumexp partials
    lse_gemm_kernel<1024, 1024><<<NB_HEAD0, 256, shbytes(1024), stream>>>(
        W0, b0, y0, 20000, pH, 0);
    lse_gemm_kernel<1024, 1024><<<1, 256, shbytes(1024), stream>>>(
        cw, cb, y0, 3, pH, NB_HEAD0);                 // cluster logits join the head
    lse_gemm_kernel<256, 256><<<NB1, 256, shbytes(256), stream>>>(
        W1, b1, y1, 20000, p1, 0);
    lse_gemm_kernel<64, 64><<<NB2, 256, shbytes(64), stream>>>(
        W2, b2, y2, 160000, p2, 0);
    lse_gemm_kernel<32, 16><<<NB3, 256, shbytes(32), stream>>>(
        W3, b3, y3, 67735, p3, 0);

    // 3) reduce partials -> logZ per cluster
    lse_reduce_kernel<<<TOKENS / 256, 256, 0, stream>>>(pH, NB_HEAD, logZ);
    lse_reduce_kernel<<<TOKENS / 256, 256, 0, stream>>>(p1, NB1, logZ + TOKENS);
    lse_reduce_kernel<<<TOKENS / 256, 256, 0, stream>>>(p2, NB2, logZ + 2 * TOKENS);
    lse_reduce_kernel<<<TOKENS / 256, 256, 0, stream>>>(p3, NB3, logZ + 3 * TOKENS);

    // 4) gathered logits + per-token logprob (one wave per token)
    gather_loss_kernel<<<(TOKENS * 32) / 256, 256, 0, stream>>>(
        target, y0, y1, y2, y3, W0, b0, W1, b1, W2, b2, W3, b3, cw, cb, logZ, loss);

    // 5) mean(-logprob)
    mean_kernel<<<1, 256, 0, stream>>>(loss, (float*)d_out);
}
